// EncoderLayer_Multiscale_linear_14542759264949
// MI455X (gfx1250) — matrix operand order repaired and checked by reading, not compiled, not run on hardware
//
#include <hip/hip_runtime.h>

// ---------------------------------------------------------------------------
// CDNA5 (gfx1250): bf16 storage, f32 accumulate; v_wmma_f32_16x16x32_bf16 for
// all channel matmuls; Tensor Data Mover (tensor_load_to_lds + s_wait_tensorcnt)
// stages K/V tiles into LDS for the linear-attention kv reduction.
// ---------------------------------------------------------------------------

typedef __bf16 bf16;
typedef __attribute__((ext_vector_type(16))) __bf16 v16bf;
typedef __attribute__((ext_vector_type(8)))  float  v8f;
typedef __attribute__((ext_vector_type(4)))  unsigned int u32x4;
typedef __attribute__((ext_vector_type(4)))  int i32x4;
typedef __attribute__((ext_vector_type(8)))  int i32x8;

__device__ inline v8f wmma_bf16(v16bf a, v16bf b, v8f c) {
  return __builtin_amdgcn_wmma_f32_16x16x32_bf16(false, a, false, b, (short)0, c,
                                                 false, false);
}

__device__ inline float gelu_exact(float x) {
  return 0.5f * x * (1.0f + erff(x * 0.70710678118654752f));
}

#define TDM_COLS 256  // columns per TDM-staged tile (16 rows x 256 cols bf16)

// Issue one TDM descriptor: load a 16 x TDM_COLS bf16 tile (row stride =
// rowStride elements in global memory) into contiguous LDS at ldsDst.
// D# layout per CDNA5 ISA (group0 + group1, 2D tensor).
__device__ inline void tdm_load_tile(bf16* ldsDst, const bf16* gsrc,
                                     int rowStride) {
  unsigned lds = (unsigned)(unsigned long long)(void*)ldsDst;
  unsigned long long ga = (unsigned long long)gsrc;
  u32x4 g0;
  g0.x = 1u;                                        // count=1, user descriptor
  g0.y = lds;                                       // lds_addr (bytes)
  g0.z = (unsigned)ga;                              // global_addr[31:0]
  g0.w = (unsigned)((ga >> 32) & 0x1FFFFFFu) | (2u << 30);  // addr[56:32]|type=2
  i32x8 g1;
  unsigned td0 = (unsigned)rowStride;               // tensor_dim0
  g1[0] = (int)(1u << 16);                          // data_size=1 (2 bytes)
  g1[1] = (int)((td0 & 0xFFFFu) << 16);             // tensor_dim0[15:0] @63:48
  g1[2] = (int)((td0 >> 16) | (16u << 16));         // dim0[31:16] | dim1=16
  g1[3] = (int)((unsigned)TDM_COLS << 16);          // tile_dim0 @127:112
  g1[4] = 16;                                       // tile_dim1=16, tile_dim2=0
  g1[5] = (int)td0;                                 // tensor_dim0_stride[31:0]
  g1[6] = 0;                                        // stride[47:32]|dim1_stride
  g1[7] = 0;
  i32x4 z4 = {0, 0, 0, 0};
#if __clang_major__ >= 23
  i32x8 z8 = {0, 0, 0, 0, 0, 0, 0, 0};
  __builtin_amdgcn_tensor_load_to_lds(g0, g1, z4, z4, z8, 0);
#else
  __builtin_amdgcn_tensor_load_to_lds(g0, g1, z4, z4, 0);
#endif
}

// ---------------------------------------------------------------------------
// f32 -> bf16 conversion
// ---------------------------------------------------------------------------
__global__ void cvt_f32_bf16(const float* __restrict__ s, bf16* __restrict__ d,
                             long n) {
  long i = (long)blockIdx.x * 256 + threadIdx.x;
  if (i < n) d[i] = (bf16)s[i];
}

// ---------------------------------------------------------------------------
// Channel GEMM: Y[b, o, l] = sum_c W[o, c] * X[b, c, l] + bias[o]
// C is a compile-time constant (128 or 256) -> fully unrolled WMMA chain.
// Fragment layouts used here (wave32):
//   A 16x32 bf16: lane m=lane&15, hi=lane>>4; elems j -> K=hi*16+j
//   B 32x16 bf16: lane n=lane&15, hi=lane>>4; elems j -> K=hi*16+j
//   D 16x16 f32 : vgpr r -> row M=r+8*hi, col N=lane&15
// ---------------------------------------------------------------------------
template <int C>
__global__ void gemm_wmma_bf16(const bf16* __restrict__ W,
                               const float* __restrict__ bias,
                               const bf16* __restrict__ X, long xBatch,
                               bf16* __restrict__ Y, long yBatch,
                               int O, int L) {
  const int lane = threadIdx.x & 31;
  const int wave = threadIdx.x >> 5;  // 0..3
  const int ltile = (blockIdx.x * 4 + wave) * 16;
  const int otile = blockIdx.y * 16;
  const int b = blockIdx.z;

  const bf16* Xb = X + (long)b * xBatch;
  bf16* Yb = Y + (long)b * yBatch;

  const int m = lane & 15;
  const int hi = lane >> 4;

  v8f acc = {};
#pragma unroll
  for (int c0 = 0; c0 < C; c0 += 32) {
    union { uint4 u[2]; v16bf v; } A;
    const bf16* wrow = W + (long)(otile + m) * C + c0 + hi * 16;
    A.u[0] = *(const uint4*)(wrow);
    A.u[1] = *(const uint4*)(wrow + 8);

    if (c0 + 32 < C)  // prefetch next K-slab of activations
      __builtin_prefetch(&Xb[(long)(c0 + 32 + hi * 16) * L + ltile + m], 0, 1);

    v16bf Bv;
    const int kbase = c0 + hi * 16;
#pragma unroll
    for (int j = 0; j < 16; ++j)
      Bv[j] = Xb[(long)(kbase + j) * L + ltile + m];

    acc = wmma_bf16(A.v, Bv, acc);
  }

  const int rowBase = hi * 8;
  const float4* bp = (const float4*)(bias + otile + rowBase);
  float4 b0 = bp[0], b1 = bp[1];
  float badd[8] = {b0.x, b0.y, b0.z, b0.w, b1.x, b1.y, b1.z, b1.w};
#pragma unroll
  for (int r = 0; r < 8; ++r) {
    int mrow = otile + rowBase + r;
    Yb[(long)mrow * L + ltile + m] = (bf16)(acc[r] + badd[r]);
  }
}

// ---------------------------------------------------------------------------
// BatchNorm stats per channel over (batch, spatial): mean + rsqrt(var + 1e-5)
// ---------------------------------------------------------------------------
__global__ void bn_stats(const bf16* __restrict__ X, long xBatch, int C, int L,
                         int B, float* __restrict__ mean,
                         float* __restrict__ rstd) {
  int c = blockIdx.x;
  int t = threadIdx.x;  // 256 threads
  float s = 0.f, s2 = 0.f;
  for (int b = 0; b < B; ++b) {
    const bf16* p = X + (long)b * xBatch + (long)c * L;
    for (int i = t; i < L; i += 256) {
      float x = (float)p[i];
      s += x;
      s2 += x * x;
    }
  }
  __shared__ float sh[512];
  sh[t] = s;
  sh[256 + t] = s2;
  __syncthreads();
  for (int k = 128; k > 0; k >>= 1) {
    if (t < k) {
      sh[t] += sh[t + k];
      sh[256 + t] += sh[256 + t + k];
    }
    __syncthreads();
  }
  if (t == 0) {
    float n = (float)B * (float)L;
    float mu = sh[0] / n;
    float var = sh[256] / n - mu * mu;
    mean[c] = mu;
    rstd[c] = rsqrtf(var + 1e-5f);
  }
}

__global__ void bn_gelu(const bf16* __restrict__ X, long xBatch,
                        bf16* __restrict__ Y, long yBatch,
                        const float* __restrict__ mean,
                        const float* __restrict__ rstd, int C, int L,
                        long total) {
  long idx = (long)blockIdx.x * 256 + threadIdx.x;
  if (idx >= total) return;
  long per = (long)C * L;
  int b = (int)(idx / per);
  long r = idx - (long)b * per;
  int c = (int)(r / L);
  long l = r - (long)c * L;
  float x = (float)X[(long)b * xBatch + (long)c * L + l];
  x = (x - mean[c]) * rstd[c];
  Y[(long)b * yBatch + (long)c * L + l] = (bf16)gelu_exact(x);
}

__global__ void relu_bf16(bf16* __restrict__ X, long n) {
  long i = (long)blockIdx.x * 256 + threadIdx.x;
  if (i < n) {
    float v = (float)X[i];
    X[i] = (bf16)fmaxf(v, 0.0f);
  }
}

// ---------------------------------------------------------------------------
// Depthwise 5x5, stride 1, SAME padding
// ---------------------------------------------------------------------------
__global__ void dwconv5(const bf16* __restrict__ X, long xBatch,
                        const float* __restrict__ Wk, bf16* __restrict__ Y,
                        long yBatch, int C, int H, int W, long total) {
  long idx = (long)blockIdx.x * 256 + threadIdx.x;
  if (idx >= total) return;
  int px = (int)(idx % W);
  long t = idx / W;
  int py = (int)(t % H);
  t /= H;
  int c = (int)(t % C);
  int b = (int)(t / C);
  const bf16* Xp = X + (long)b * xBatch + (long)c * H * W;
  const float* wp = Wk + c * 25;
  float acc = 0.f;
#pragma unroll
  for (int ky = 0; ky < 5; ++ky) {
    int iy = py + ky - 2;
    if (iy < 0 || iy >= H) continue;
#pragma unroll
    for (int kx = 0; kx < 5; ++kx) {
      int ix = px + kx - 2;
      if (ix < 0 || ix >= W) continue;
      acc += (float)Xp[(long)iy * W + ix] * wp[ky * 5 + kx];
    }
  }
  Y[(long)b * yBatch + (long)c * H * W + (long)py * W + px] = (bf16)acc;
}

// ---------------------------------------------------------------------------
// Linear attention phase 1 (TDM + WMMA): per (b, head)
//   kv[d, dv] = (1/L) * sum_s K[d, s] * V[dv, s]
//   ksum[d]   = (1/L) * sum_s K[d, s]
// K/V 16xL slabs are staged 16x256 at a time into LDS by the Tensor Data
// Mover (wave 0 issues the descriptors, s_wait_tensorcnt, barrier), then the
// 8 waves each run one 16x16x32 WMMA on their 32-column sub-tile.
// ---------------------------------------------------------------------------
__global__ void attn_kv(const bf16* __restrict__ Kc, const bf16* __restrict__ Vc,
                        float* __restrict__ kv, float* __restrict__ ksum,
                        int L) {
  __shared__ __align__(16) bf16 shK[16 * TDM_COLS];
  __shared__ __align__(16) bf16 shV[16 * TDM_COLS];
  __shared__ float red[8 * 256];

  int h = blockIdx.x;
  int b = blockIdx.y;
  const bf16* Kh = Kc + ((long)b * 256 + h * 16) * L;
  const bf16* Vh = Vc + ((long)b * 256 + h * 16) * L;
  int lane = threadIdx.x & 31, wave = threadIdx.x >> 5;  // 8 waves
  int m = lane & 15, hi = lane >> 4;

  v8f acc = {};
  for (int c0 = 0; c0 < L; c0 += TDM_COLS) {
    __syncthreads();  // previous tile fully consumed
    if (wave == 0) {
      tdm_load_tile(shK, Kh + c0, L);
      tdm_load_tile(shV, Vh + c0, L);
      __builtin_amdgcn_s_wait_tensorcnt(0);
    }
    __syncthreads();

    int s0 = wave * 32;  // this wave's 32-column sub-tile
    union { uint4 u[2]; v16bf v; } A;  // A[m,k] = K[m, c0+s0+k]
    const bf16* arow = shK + m * TDM_COLS + s0 + hi * 16;
    A.u[0] = *(const uint4*)(arow);
    A.u[1] = *(const uint4*)(arow + 8);
    union { uint4 u[2]; v16bf v; } Bf;  // B[k,n] = V[n, c0+s0+k]
    const bf16* brow = shV + m * TDM_COLS + s0 + hi * 16;
    Bf.u[0] = *(const uint4*)(brow);
    Bf.u[1] = *(const uint4*)(brow + 8);
    acc = wmma_bf16(A.v, Bf.v, acc);
  }

  int rowBase = hi * 8;
#pragma unroll
  for (int r = 0; r < 8; ++r)
    red[wave * 256 + (rowBase + r) * 16 + m] = acc[r];
  __syncthreads();

  const float invL = 1.0f / (float)L;
  float s = 0.f;
  for (int w = 0; w < 8; ++w) s += red[w * 256 + threadIdx.x];
  kv[((long)b * 16 + h) * 256 + threadIdx.x] = s * invL;
  __syncthreads();

  // ksum: 16 chunks x 16 d
  int d = threadIdx.x & 15, chunk = threadIdx.x >> 4;
  float ks = 0.f;
  for (int si = chunk; si < L; si += 16) ks += (float)Kh[(long)d * L + si];
  red[chunk * 16 + d] = ks;
  __syncthreads();
  if (threadIdx.x < 16) {
    float tot = 0.f;
    for (int cp = 0; cp < 16; ++cp) tot += red[cp * 16 + threadIdx.x];
    ksum[((long)b * 16 + h) * 16 + threadIdx.x] = tot * invL;
  }
}

// ---------------------------------------------------------------------------
// Linear attention phase 2: per 16-position tile per head
//   z[l]      = 1 / max(qf[l,:] . ksum, 1e-6)
//   out[l,dv] = (sum_d qf[l,d] * kv[d,dv]) * z[l]     (one WMMA, K padded)
// ---------------------------------------------------------------------------
__global__ void attn_apply(const bf16* __restrict__ Qc,
                           const float* __restrict__ kv,
                           const float* __restrict__ ksum,
                           bf16* __restrict__ Out, int L) {
  int h = blockIdx.y, b = blockIdx.z;
  int lane = threadIdx.x & 31, wave = threadIdx.x >> 5;  // 4 waves
  int ltile = (blockIdx.x * 4 + wave) * 16;
  const bf16* Qh = Qc + ((long)b * 256 + h * 16) * L;
  const float* kvh = kv + ((long)b * 16 + h) * 256;
  const float* ksh = ksum + ((long)b * 16 + h) * 16;
  int m = lane & 15, hi = lane >> 4;

  float dot = 0.f;
#pragma unroll
  for (int d = 0; d < 16; ++d)
    dot += (float)Qh[(long)d * L + ltile + m] * ksh[d];
  float z = 1.0f / fmaxf(dot, 1e-6f);

  v16bf Av;  // A[m,k] = qf[ltile+m, k] for k<16, zero above
#pragma unroll
  for (int j = 0; j < 8; ++j) {
    int k0 = hi * 8 + j;  // 0..15
    Av[j] = Qh[(long)k0 * L + ltile + m];
    Av[8 + j] = (bf16)0.0f;  // K = 16..31
  }
  v16bf Bv;  // B[k,n] = kv[k][n] for k<16 (elems 0..7, k=hi*8+j), zero for K>=16
#pragma unroll
  for (int j = 0; j < 16; ++j) {
    float v = (j < 8) ? kvh[(hi * 8 + j) * 16 + m] : 0.0f;
    Bv[j] = (bf16)v;
  }
  v8f acc = {};
  acc = wmma_bf16(Av, Bv, acc);

  int rowBase = hi * 8;
#pragma unroll
  for (int r = 0; r < 8; ++r) {
    float zr = __shfl(z, rowBase + r, 32);
    Out[((long)b * 256 + h * 16 + m) * L + ltile + rowBase + r] =
        (bf16)(acc[r] * zr);
  }
}

// ---------------------------------------------------------------------------
// LayerNorm over channels of (f32 base + bf16 addend), write bf16
// ---------------------------------------------------------------------------
__global__ void ln_add_f32base(const float* __restrict__ Xf,
                               const bf16* __restrict__ Add, long addBatch,
                               const float* __restrict__ w,
                               const float* __restrict__ bia,
                               bf16* __restrict__ Y, long yBatch, int C, int L,
                               long total) {
  long idx = (long)blockIdx.x * 256 + threadIdx.x;
  if (idx >= total) return;
  int b = (int)(idx / L);
  long l = idx - (long)b * L;
  const float* xp = Xf + (long)b * C * L + l;
  const bf16* ap = Add + (long)b * addBatch + l;
  float s = 0.f, s2 = 0.f;
  for (int c = 0; c < C; ++c) {
    float v = xp[(long)c * L] + (float)ap[(long)c * L];
    s += v;
    s2 += v * v;
  }
  float mu = s / (float)C;
  float var = s2 / (float)C - mu * mu;
  float rs = rsqrtf(var + 1e-5f);
  for (int c = 0; c < C; ++c) {
    float v = xp[(long)c * L] + (float)ap[(long)c * L];
    Y[(long)b * yBatch + (long)c * L + l] = (bf16)((v - mu) * rs * w[c] + bia[c]);
  }
}

// LayerNorm over channels of (bf16 + bf16), write f32 (final output)
__global__ void ln_add_out(const bf16* __restrict__ A, long aBatch,
                           const bf16* __restrict__ Bd, long bBatch,
                           const float* __restrict__ w,
                           const float* __restrict__ bia,
                           float* __restrict__ Y, int C, int L, long total) {
  long idx = (long)blockIdx.x * 256 + threadIdx.x;
  if (idx >= total) return;
  int b = (int)(idx / L);
  long l = idx - (long)b * L;
  const bf16* ap = A + (long)b * aBatch + l;
  const bf16* bp = Bd + (long)b * bBatch + l;
  float s = 0.f, s2 = 0.f;
  for (int c = 0; c < C; ++c) {
    float v = (float)ap[(long)c * L] + (float)bp[(long)c * L];
    s += v;
    s2 += v * v;
  }
  float mu = s / (float)C;
  float var = s2 / (float)C - mu * mu;
  float rs = rsqrtf(var + 1e-5f);
  for (int c = 0; c < C; ++c) {
    float v = (float)ap[(long)c * L] + (float)bp[(long)c * L];
    Y[(long)b * C * L + (long)c * L + l] = (v - mu) * rs * w[c] + bia[c];
  }
}

// ---------------------------------------------------------------------------
// Host orchestration
// ---------------------------------------------------------------------------
extern "C" void kernel_launch(void* const* d_in, const int* in_sizes, int n_in,
                              void* d_out, int out_size, void* d_ws,
                              size_t ws_size, hipStream_t stream) {
  const int Bt = 2, D = 128, Hh = 160, Ww = 160;
  const long L = (long)Hh * Ww;        // 25600
  const long S1 = (long)Bt * D * L;    // [B,128,L] elems
  const long S2 = (long)Bt * 2 * D * L;

  const float* x      = (const float*)d_in[0];
  const float* source = (const float*)d_in[1];
  const float* wq = (const float*)d_in[2];  const float* bq = (const float*)d_in[3];
  const float* wk = (const float*)d_in[4];  const float* bk = (const float*)d_in[5];
  const float* wv = (const float*)d_in[6];  const float* bv = (const float*)d_in[7];
  const float* dwq = (const float*)d_in[8];  const float* pwq = (const float*)d_in[9];
  const float* dwk = (const float*)d_in[10]; const float* pwk = (const float*)d_in[11];
  const float* dwv = (const float*)d_in[12]; const float* pwv = (const float*)d_in[13];
  const float* wproj  = (const float*)d_in[14];
  const float* wmerge = (const float*)d_in[15]; const float* bmerge = (const float*)d_in[16];
  const float* wff1 = (const float*)d_in[17]; const float* bff1 = (const float*)d_in[18];
  const float* wff2 = (const float*)d_in[19]; const float* bff2 = (const float*)d_in[20];
  const float* ln1w = (const float*)d_in[21]; const float* ln1b = (const float*)d_in[22];
  const float* ln2w = (const float*)d_in[23]; const float* ln2b = (const float*)d_in[24];
  float* out = (float*)d_out;

  char* wp = (char*)d_ws;
  auto alloc = [&](size_t bytes) -> char* {
    char* r = wp;
    wp += (bytes + 255) & ~(size_t)255;
    return r;
  };

  bf16* xb = (bf16*)alloc(S1 * 2);
  bf16* sb = (bf16*)alloc(S1 * 2);
  bf16* qc = (bf16*)alloc(S2 * 2);
  bf16* kc = (bf16*)alloc(S2 * 2);
  bf16* vc = (bf16*)alloc(S2 * 2);
  bf16* oc = (bf16*)alloc(S2 * 2);   // attention output, later ff1 activations
  bf16* t0 = (bf16*)alloc(S1 * 2);   // pre-activation scratch
  bf16* t1 = (bf16*)alloc(S1 * 2);   // dwconv out / merge-gelu
  bf16* m0 = (bf16*)alloc(S1 * 2);   // msg after wproj
  bf16* m1 = (bf16*)alloc(S1 * 2);   // msg after ln1 (residual for ln2)
  bf16* m2 = (bf16*)alloc(S1 * 2);   // ff2 gelu out
  bf16* wqb = (bf16*)alloc(D * D * 2);
  bf16* wkb = (bf16*)alloc(D * D * 2);
  bf16* wvb = (bf16*)alloc(D * D * 2);
  bf16* pwqb = (bf16*)alloc(D * D * 2);
  bf16* pwkb = (bf16*)alloc(D * D * 2);
  bf16* pwvb = (bf16*)alloc(D * D * 2);
  bf16* wprojb = (bf16*)alloc(D * 2 * D * 2);
  bf16* wmergeb = (bf16*)alloc(D * D * 2);
  bf16* wff1b = (bf16*)alloc(2 * D * D * 2);
  bf16* wff2b = (bf16*)alloc(D * 2 * D * 2);
  float* meanBuf = (float*)alloc(256 * 4);
  float* rstdBuf = (float*)alloc(256 * 4);
  float* zbias = (float*)alloc(256 * 4);
  float* kvBuf = (float*)alloc((size_t)Bt * 16 * 256 * 4);
  float* ksumBuf = (float*)alloc((size_t)Bt * 16 * 16 * 4);

  hipMemsetAsync(zbias, 0, 256 * sizeof(float), stream);

  auto cvt = [&](const float* s, bf16* d, long n) {
    cvt_f32_bf16<<<(unsigned)((n + 255) / 256), 256, 0, stream>>>(s, d, n);
  };
  auto gemm = [&](const bf16* W, const float* bias, const bf16* X, long xB,
                  bf16* Y, long yB, int O, int C) {
    dim3 g((unsigned)(L / 64), (unsigned)(O / 16), (unsigned)Bt);
    const float* bp = bias ? bias : zbias;
    if (C == 128)
      gemm_wmma_bf16<128><<<g, 128, 0, stream>>>(W, bp, X, xB, Y, yB, O, (int)L);
    else
      gemm_wmma_bf16<256><<<g, 128, 0, stream>>>(W, bp, X, xB, Y, yB, O, (int)L);
  };
  auto bn = [&](const bf16* X, long xB, bf16* Y, long yB, int C) {
    bn_stats<<<C, 256, 0, stream>>>(X, xB, C, (int)L, Bt, meanBuf, rstdBuf);
    long total = (long)Bt * C * L;
    bn_gelu<<<(unsigned)(total / 256), 256, 0, stream>>>(X, xB, Y, yB, meanBuf,
                                                         rstdBuf, C, (int)L,
                                                         total);
  };
  auto dw = [&](const bf16* X, long xB, const float* Wk, bf16* Y, long yB) {
    long total = (long)Bt * D * L;
    dwconv5<<<(unsigned)(total / 256), 256, 0, stream>>>(X, xB, Wk, Y, yB, D,
                                                         Hh, Ww, total);
  };
  // q/k/v branch: GEMM+bias -> BN+GELU (into first 128 channels of dst) ->
  // depthwise 5x5 -> pointwise GEMM (into second 128 channels of dst)
  auto branch = [&](const bf16* inp, const bf16* Wm, const float* bias,
                    const float* dwW, const bf16* pwW, bf16* dst) {
    gemm(Wm, bias, inp, D * L, t0, D * L, D, D);
    bn(t0, D * L, dst, 2 * D * L, D);
    dw(dst, 2 * D * L, dwW, t1, D * L);
    gemm(pwW, nullptr, t1, D * L, dst + (long)D * L, 2 * D * L, D, D);
  };

  // --- precision conversions ---
  cvt(x, xb, S1);
  cvt(source, sb, S1);
  cvt(wq, wqb, D * D);       cvt(wk, wkb, D * D);       cvt(wv, wvb, D * D);
  cvt(pwq, pwqb, D * D);     cvt(pwk, pwkb, D * D);     cvt(pwv, pwvb, D * D);
  cvt(wproj, wprojb, (long)D * 2 * D);
  cvt(wmerge, wmergeb, D * D);
  cvt(wff1, wff1b, (long)2 * D * D);
  cvt(wff2, wff2b, (long)D * 2 * D);

  // --- q/k/v multiscale branches ---
  branch(xb, wqb, bq, dwq, pwqb, qc);
  relu_bf16<<<(unsigned)(S2 / 256), 256, 0, stream>>>(qc, S2);
  branch(sb, wkb, bk, dwk, pwkb, kc);
  relu_bf16<<<(unsigned)(S2 / 256), 256, 0, stream>>>(kc, S2);
  branch(sb, wvb, bv, dwv, pwvb, vc);  // vc: no relu

  // --- linear attention ---
  attn_kv<<<dim3(16, Bt), 256, 0, stream>>>(kc, vc, kvBuf, ksumBuf, (int)L);
  attn_apply<<<dim3((unsigned)(L / 64), 16, Bt), 128, 0, stream>>>(
      qc, kvBuf, ksumBuf, oc, (int)L);

  // --- projection + merge + residual + LN1 ---
  gemm(wprojb, nullptr, oc, 2 * D * L, m0, D * L, D, 2 * D);
  gemm(wmergeb, bmerge, m0, D * L, t0, D * L, D, D);
  bn(t0, D * L, t1, D * L, D);
  long lnTot = (long)Bt * L;
  ln_add_f32base<<<(unsigned)(lnTot / 256), 256, 0, stream>>>(
      x, t1, D * L, ln1w, ln1b, m1, D * L, D, (int)L, lnTot);

  // --- conv FFN + residual + LN2 ---
  gemm(wff1b, bff1, m1, D * L, oc, 2 * D * L, 2 * D, D);
  bn(oc, 2 * D * L, oc, 2 * D * L, 2 * D);
  gemm(wff2b, bff2, oc, 2 * D * L, t0, D * L, D, 2 * D);
  bn(t0, D * L, m2, D * L, D);
  ln_add_out<<<(unsigned)(lnTot / 256), 256, 0, stream>>>(
      m2, D * L, m1, D * L, ln2w, ln2b, out, D, (int)L, lnTot);
}
